// TFNO4DEncoder_5214090297578
// MI455X (gfx1250) — matrix operand order repaired and checked by reading, not compiled, not run on hardware
//
#include <hip/hip_runtime.h>
#include <hip/hip_bf16.h>
#include <math.h>

// ---------------------------------------------------------------------------
// TFNO4D encoder forward for gfx1250 (MI455X).
//   grid: padded N=32 per dim, C=O=32 channels, rfft half-dim H=17.
// ---------------------------------------------------------------------------

typedef __attribute__((ext_vector_type(16))) _Float16 v16h;
typedef __attribute__((ext_vector_type(8)))  float    v8f;
typedef __attribute__((ext_vector_type(4)))  unsigned u32x4;
typedef __attribute__((ext_vector_type(8)))  int      i32x8;
typedef __attribute__((ext_vector_type(4)))  int      i32x4;

#define NPAD   32
#define NP     (1L << 20)          // 32^4 points
#define HFREQ  17
#define SPEC_PER_C 557056L         // 32*32*32*17
#define NPTS24 331776              // 24^4

#if __has_builtin(__builtin_amdgcn_tensor_load_to_lds) && \
    __has_builtin(__builtin_amdgcn_s_wait_tensorcnt)
#define HAVE_TDM 1
#endif

__device__ inline float gelu_exact(float v) {
    return 0.5f * v * (1.0f + erff(v * 0.70710678118654752f));
}

__device__ inline float2 cmulf(float2 a, float2 b) {
    return make_float2(a.x * b.x - a.y * b.y, a.x * b.y + a.y * b.x);
}

// 32-point complex FFT across the 32 lanes of a wave (one element per lane).
// Radix-2 DIF with __shfl_xor butterflies + bit-reversal unscramble.
// sgn = -1 : forward (exp(-i...)),  sgn = +1 : inverse (unscaled).
__device__ inline float2 fft32(float2 v, int lane, float sgn) {
    #pragma unroll
    for (int m = 16; m >= 1; m >>= 1) {
        float ox = __shfl_xor(v.x, m, 32);
        float oy = __shfl_xor(v.y, m, 32);
        int j = lane & (m - 1);
        if (lane & m) {
            float ang = sgn * 3.14159265358979323846f * (float)j / (float)m;
            float s, c;
            __sincosf(ang, &s, &c);
            float2 d = make_float2(ox - v.x, oy - v.y);
            v = cmulf(d, make_float2(c, s));
        } else {
            v.x += ox;
            v.y += oy;
        }
    }
    int r = __brev((unsigned)lane) >> 27;   // 5-bit bit reversal
    v.x = __shfl(v.x, r, 32);
    v.y = __shfl(v.y, r, 32);
    return v;
}

// ---------------------------------------------------------------------------
// Lift: coords + 7->16 GELU -> 32, write into zeroed padded buffer.
// ---------------------------------------------------------------------------
__global__ void k_lift(const float* __restrict__ xin,
                       const float* __restrict__ lw1, const float* __restrict__ lb1,
                       const float* __restrict__ lw2, const float* __restrict__ lb2,
                       float* __restrict__ out) {
    int p = blockIdx.x * blockDim.x + threadIdx.x;
    if (p >= NPTS24) return;
    int t =  p        % 24;
    int z = (p / 24)  % 24;
    int y = (p / 576) % 24;
    int x =  p / 13824;
    float f[7];
    f[0] = xin[p];
    f[1] = xin[NPTS24 + p];
    f[2] = xin[2 * NPTS24 + p];
    f[3] = (float)x * (1.0f / 23.0f);
    f[4] = (float)y * (1.0f / 23.0f);
    f[5] = (float)z * (1.0f / 23.0f);
    f[6] = (float)t * (1.0f / 23.0f);
    float h[16];
    #pragma unroll
    for (int j = 0; j < 16; ++j) {
        float a = lb1[j];
        #pragma unroll
        for (int i = 0; i < 7; ++i) a += lw1[j * 7 + i] * f[i];
        h[j] = gelu_exact(a);
    }
    long op = (long)(((x * NPAD + y) * NPAD + z) * NPAD + t);
    for (int c = 0; c < 32; ++c) {
        float a = lb2[c];
        #pragma unroll
        for (int j = 0; j < 16; ++j) a += lw2[c * 16 + j] * h[j];
        out[(long)c * NP + op] = a;
    }
}

// ---------------------------------------------------------------------------
// Pointwise conv y[o,p] = W[o,c] x[c,p] + b[o] via v_wmma_f32_16x16x32_f16.
// W is staged global->LDS by the Tensor Data Mover (one wave issues the TDM
// descriptor, waits on TENSORcnt), then converted to f16 for the A operand.
// Each wave keeps both 16x32 A tiles in registers and sweeps 64 points
// (4 B tiles) -> 8 WMMAs per wave with full A reuse.
// ---------------------------------------------------------------------------
__global__ void k_pointwise(const float* __restrict__ xin, float* __restrict__ yout,
                            const float* __restrict__ W, const float* __restrict__ bias) {
    __shared__ float    Wstage[1024];
    __shared__ _Float16 Ws[32][32];
    __shared__ float    bs[32];
    int tid  = threadIdx.x;
    int wid  = tid >> 5;
    int lane = tid & 31;

#ifdef HAVE_TDM
    if (wid == 0) {
        // Tensor DMA descriptor: 1-D tensor, 1024 x f32, global -> LDS.
        unsigned long long ga = (unsigned long long)(const void*)W;
        unsigned ldsa = (unsigned)(unsigned long long)(void*)Wstage;
        u32x4 g0;
        g0[0] = 1u;                                            // count=1 (valid user D#)
        g0[1] = ldsa;                                          // lds_addr
        g0[2] = (unsigned)(ga & 0xffffffffu);                  // global_addr[31:0]
        g0[3] = (unsigned)((ga >> 32) & 0x01ffffffu)           // global_addr[56:32]
                | (2u << 30);                                  // type=2 (image)
        i32x8 g1;
        g1[0] = (int)(2u << 16);        // workgroup_mask=0, data_size=2 (4 bytes)
        g1[1] = (int)(1024u << 16);     // tensor_dim0 low16 in [63:48]
        g1[2] = 0;                      // tensor_dim0 hi16 = 0, tensor_dim1 = 0
        g1[3] = (int)(1024u << 16);     // tile_dim0 = 1024 in [127:112]
        g1[4] = 0;                      // tile_dim1 = tile_dim2 = 0 (unused)
        g1[5] = 1024;                   // tensor_dim0_stride low32
        g1[6] = 0;
        g1[7] = 0;
        i32x4 z4 = {0, 0, 0, 0};
        i32x8 z8 = {0, 0, 0, 0, 0, 0, 0, 0};
        __builtin_amdgcn_tensor_load_to_lds(g0, g1, z4, z4, z8, 0);
        __builtin_amdgcn_s_wait_tensorcnt(0);
    }
#else
    for (int i = tid; i < 1024; i += 128) Wstage[i] = W[i];
#endif
    if (tid < 32) bs[tid] = bias[tid];
    __syncthreads();
    for (int i = tid; i < 1024; i += 128) Ws[i >> 5][i & 31] = (_Float16)Wstage[i];
    __syncthreads();

    int g = lane >> 4;          // lane half-group
    int m = lane & 15;
    long p0 = (long)(blockIdx.x * 4 + wid) * 64;

    // A matrices (16 x 32 f16), kept resident for the whole point sweep.
    // lane row M=m; VGPR0-3 hold K=0..7 (+8g), VGPR4-7 hold K=16..23 (+8g).
    v16h a0, a1;
    #pragma unroll
    for (int e = 0; e < 16; ++e) {
        int k = (e < 8) ? (e + 8 * g) : (e + 8 + 8 * g);
        a0[e] = Ws[m][k];
        a1[e] = Ws[16 + m][k];
    }

    #pragma unroll
    for (int s = 0; s < 4; ++s) {
        long ps = p0 + s * 16;
        // B matrix (32 x 16): lane col N=m; VGPR0-3 hold K=0..7 (+8g), VGPR4-7 hold K=16..23 (+8g)
        v16h b;
        #pragma unroll
        for (int e = 0; e < 16; ++e) {
            int k = (e < 8) ? (e + 8 * g) : (e + 8 + 8 * g);   // channel
            b[e] = (_Float16)xin[(long)k * NP + ps + m];
        }
        v8f c0 = {}, c1 = {};
        c0 = __builtin_amdgcn_wmma_f32_16x16x32_f16(false, a0, false, b,
                                                    (short)0, c0, false, false);
        c1 = __builtin_amdgcn_wmma_f32_16x16x32_f16(false, a1, false, b,
                                                    (short)0, c1, false, false);
        #pragma unroll
        for (int r = 0; r < 8; ++r) {
            int o0 = r + 8 * g;
            int o1 = 16 + r + 8 * g;
            yout[(long)o0 * NP + ps + m] = c0[r] + bs[o0];
            yout[(long)o1 * NP + ps + m] = c1[r] + bs[o1];
        }
    }
}

// ---------------------------------------------------------------------------
// rFFT along T: one wave per line of 32 reals -> 17 complex bins.
// ---------------------------------------------------------------------------
__global__ void k_rfft_t(const float* __restrict__ xin, float2* __restrict__ spec) {
    int wid  = threadIdx.x >> 5;
    int lane = threadIdx.x & 31;
    int line = blockIdx.x * 8 + wid;       // < 32 * 32768
    int c  = line >> 15;
    int sp = line & 32767;                 // (x*32+y)*32+z
    float2 v = make_float2(xin[((long)c << 20) + ((long)sp << 5) + lane], 0.0f);
    v = fft32(v, lane, -1.0f);
    if (lane < HFREQ)
        spec[((long)c * 32768 + sp) * HFREQ + lane] = v;
}

// ---------------------------------------------------------------------------
// Complex FFT along X/Y/Z of spec[c][x][y][z][h]; in place, per-wave lines.
// ---------------------------------------------------------------------------
__global__ void k_fft_axis(float2* __restrict__ spec, int axis, float sgn, float scale) {
    int wid  = threadIdx.x >> 5;
    int lane = threadIdx.x & 31;
    int line = blockIdx.x * 8 + wid;       // < 32*32*32*17 = 557056
    int h  = line % HFREQ;
    int q  = line / HFREQ;
    int b_ = q & 31;
    int a_ = (q >> 5) & 31;
    int c  = q >> 10;
    long base, stride;
    if (axis == 0) {        // vary X; a_=y, b_=z
        base = ((((long)c * 32) * 32 + a_) * 32 + b_) * HFREQ + h;
        stride = 32L * 32 * HFREQ;
    } else if (axis == 1) { // vary Y; a_=x, b_=z
        base = ((((long)c * 32 + a_) * 32) * 32 + b_) * HFREQ + h;
        stride = 32L * HFREQ;
    } else {                // vary Z; a_=x, b_=y
        base = ((((long)c * 32 + a_) * 32 + b_) * 32) * HFREQ + h;
        stride = HFREQ;
    }
    float2 v = spec[base + (long)lane * stride];
    v = fft32(v, lane, sgn);
    v.x *= scale;
    v.y *= scale;
    spec[base + (long)lane * stride] = v;
}

// ---------------------------------------------------------------------------
// Spectral mode multiply, in place: out[o] = sum_c w[c,o] * xft[c].
// One wave per kept mode point; lane = output channel.
// ---------------------------------------------------------------------------
__global__ void k_modemul(float2* __restrict__ spec, const float* __restrict__ w) {
    __shared__ float2 xs[8][32];
    int wid  = threadIdx.x >> 5;
    int lane = threadIdx.x & 31;
    int mp = blockIdx.x * 8 + wid;         // < 32768 (8 corners * 8^4 modes)
    int corner = mp >> 12;
    int i1 = (mp >> 9) & 7;
    int i2 = (mp >> 6) & 7;
    int i3 = (mp >> 3) & 7;
    int i4 =  mp       & 7;
    int fx = (corner & 4) ? 24 + i1 : i1;
    int fy = (corner & 2) ? 24 + i2 : i2;
    int fz = (corner & 1) ? 24 + i3 : i3;
    long sbase = (((long)fx * 32 + fy) * 32 + fz) * HFREQ + i4;

    xs[wid][lane] = spec[(long)lane * SPEC_PER_C + sbase];
    __syncthreads();

    int mode = ((i1 * 8 + i2) * 8 + i3) * 8 + i4;
    // w layout: [8][C=32][O=32][4096][2]
    long wbase = (((long)corner * 32 * 32 + lane) * 4096 + mode) * 2;
    const long cstride = 32L * 4096 * 2;
    float2 acc = make_float2(0.0f, 0.0f);
    #pragma unroll 4
    for (int c = 0; c < 32; ++c) {
        if (c + 8 < 32)
            __builtin_prefetch(&w[wbase + (long)(c + 8) * cstride], 0, 1);
        float wr = w[wbase + (long)c * cstride];
        float wi = w[wbase + (long)c * cstride + 1];
        float2 xv = xs[wid][c];
        acc.x = fmaf(wr, xv.x, fmaf(-wi, xv.y, acc.x));
        acc.y = fmaf(wr, xv.y, fmaf( wi, xv.x, acc.y));
    }
    spec[(long)lane * SPEC_PER_C + sbase] = acc;
}

// Zero every frequency bin outside the kept mode corners.
__global__ void k_zeromodes(float2* __restrict__ spec) {
    long i = (long)blockIdx.x * 256 + threadIdx.x;
    if (i >= 32L * SPEC_PER_C) return;
    int h = (int)(i % HFREQ);
    long q = i / HFREQ;
    int z = (int)(q & 31);
    int y = (int)((q >> 5) & 31);
    int x = (int)((q >> 10) & 31);
    bool keep = (h < 8) && ((x < 8) || (x >= 24))
                        && ((y < 8) || (y >= 24))
                        && ((z < 8) || (z >= 24));
    if (!keep) spec[i] = make_float2(0.0f, 0.0f);
}

// ---------------------------------------------------------------------------
// irFFT along T (Hermitian reconstruct) and accumulate into y.
// ---------------------------------------------------------------------------
__global__ void k_irfft_t_add(const float2* __restrict__ spec, float* __restrict__ y) {
    int wid  = threadIdx.x >> 5;
    int lane = threadIdx.x & 31;
    int line = blockIdx.x * 8 + wid;
    int c  = line >> 15;
    int sp = line & 32767;
    int h  = (lane < HFREQ) ? lane : 32 - lane;
    float2 v = spec[((long)c * 32768 + sp) * HFREQ + h];
    if (lane > 16) v.y = -v.y;             // conjugate mirror
    v = fft32(v, lane, +1.0f);
    y[((long)c << 20) + ((long)sp << 5) + lane] += v.x * (1.0f / 32.0f);
}

__global__ void k_gelu(const float* __restrict__ yin, float* __restrict__ xout, long n) {
    long i = (long)blockIdx.x * 256 + threadIdx.x;
    if (i < n) xout[i] = gelu_exact(yin[i]);
}

__global__ void k_crop(const float* __restrict__ y, float* __restrict__ out) {
    long i = (long)blockIdx.x * 256 + threadIdx.x;
    if (i >= 32L * NPTS24) return;
    long p = i % NPTS24;
    int c  = (int)(i / NPTS24);
    int t  = (int)(p % 24);
    long q = p / 24;
    int z  = (int)(q % 24); q /= 24;
    int yy = (int)(q % 24);
    int x  = (int)(q / 24);
    out[i] = y[((long)c << 20) + (long)(((x * NPAD + yy) * NPAD + z) * NPAD + t)];
}

// ---------------------------------------------------------------------------
extern "C" void kernel_launch(void* const* d_in, const int* in_sizes, int n_in,
                              void* d_out, int out_size, void* d_ws, size_t ws_size,
                              hipStream_t stream) {
    const float* xin    = (const float*)d_in[0];
    const float* lw1    = (const float*)d_in[1];
    const float* lb1    = (const float*)d_in[2];
    const float* lw2    = (const float*)d_in[3];
    const float* lb2    = (const float*)d_in[4];
    const float* conv_w = (const float*)d_in[5];
    const float* conv_b = (const float*)d_in[6];
    const float* spec_w = (const float*)d_in[7];

    float*  bufA = (float*)d_ws;                     // activations (padded)
    float*  bufB = bufA + 32L * NP;                  // layer output y
    float2* spec = (float2*)(bufB + 32L * NP);       // rfft spectrum

    // zero padding region of the initial activation buffer
    (void)hipMemsetAsync(bufA, 0, 32L * NP * sizeof(float), stream);

    k_lift<<<NPTS24 / 256, 256, 0, stream>>>(xin, lw1, lb1, lw2, lb2, bufA);

    const long spec_layer = 8L * 32 * 32 * 4096 * 2;
    for (int k = 0; k < 4; ++k) {
        k_pointwise<<<4096, 128, 0, stream>>>(bufA, bufB,
                                              conv_w + (long)k * 1024,
                                              conv_b + (long)k * 32);
        // forward 4D rFFT of bufA
        k_rfft_t<<<131072, 256, 0, stream>>>(bufA, spec);
        k_fft_axis<<<69632, 256, 0, stream>>>(spec, 2, -1.0f, 1.0f);
        k_fft_axis<<<69632, 256, 0, stream>>>(spec, 1, -1.0f, 1.0f);
        k_fft_axis<<<69632, 256, 0, stream>>>(spec, 0, -1.0f, 1.0f);
        // mode multiply (in place) + zero non-kept bins
        k_modemul<<<4096, 256, 0, stream>>>(spec, spec_w + (long)k * spec_layer);
        k_zeromodes<<<69632, 256, 0, stream>>>(spec);
        // inverse 4D FFT, accumulate into bufB
        k_fft_axis<<<69632, 256, 0, stream>>>(spec, 0, +1.0f, 1.0f / 32.0f);
        k_fft_axis<<<69632, 256, 0, stream>>>(spec, 1, +1.0f, 1.0f / 32.0f);
        k_fft_axis<<<69632, 256, 0, stream>>>(spec, 2, +1.0f, 1.0f / 32.0f);
        k_irfft_t_add<<<131072, 256, 0, stream>>>(spec, bufB);

        if (k < 3)
            k_gelu<<<131072, 256, 0, stream>>>(bufB, bufA, 32L * NP);
    }
    k_crop<<<(32L * NPTS24 + 255) / 256, 256, 0, stream>>>(bufB, (float*)d_out);
}
